// MIND_42090679501344
// MI455X (gfx1250) — compile-verified, matrix operand order checked by reading, not run on hardware
//
#include <hip/hip_runtime.h>

// ---------------- problem constants ----------------
constexpr int kB = 1024;      // batch
constexpr int kS = 200;       // seq len
constexpr int kI = 4;         // capsules
constexpr int kH = 64;        // hidden
constexpr int kN = 100000;    // items
constexpr int kNSPLIT = 25;   // N-splits for scores GEMM
constexpr int kTILES = 250;   // 16-col tiles per split (25*250*16 = 100000)

// ---------------- CDNA5 WMMA types ----------------
typedef __attribute__((ext_vector_type(16))) __bf16 v16bf;
typedef __attribute__((ext_vector_type(8)))  float  v8f;

union BfVec { v16bf v; uint4 q[2]; };

__device__ __forceinline__ unsigned short f2bf(float f) {
  unsigned u = __float_as_uint(f);
  unsigned r = (u + 0x7FFFu + ((u >> 16) & 1u)) >> 16;   // round-to-nearest-even
  return (unsigned short)r;
}

// one 16x16 score tile: D = A(16x64) x B(64x16) via two chained bf16 WMMAs
__device__ __forceinline__ v8f score_tile(const uint4* __restrict__ b4, int o,
                                          const BfVec& alo, const BfVec& ahi) {
  BfVec blo, bhi;
  blo.q[0] = b4[o];     blo.q[1] = b4[o + 2];
  bhi.q[0] = b4[o + 4]; bhi.q[1] = b4[o + 6];
  v8f c = {0.f, 0.f, 0.f, 0.f, 0.f, 0.f, 0.f, 0.f};
  c = __builtin_amdgcn_wmma_f32_16x16x32_bf16(false, alo.v, false, blo.v, (short)0, c, false, false);
  c = __builtin_amdgcn_wmma_f32_16x16x32_bf16(false, ahi.v, false, bhi.v, (short)0, c, false, false);
  return c;
}

// ---------------- K1: hat[b,s,:] = item_emb[item_seq[b,s]] @ W^T ----------------
// block = 256 threads = 4 rows x 64 outputs; W (16KB) stays in WGP$ across blocks.
__global__ void k_hat(const int* __restrict__ item_seq,
                      const float* __restrict__ item_emb,
                      const float* __restrict__ W,
                      float* __restrict__ hat) {
  __shared__ __align__(16) float eL[4 * 64];
  const int tid = threadIdx.x;
  const int r0  = blockIdx.x * 4;
  const int rl  = tid >> 6;
  const int h   = tid & 63;
  const int idx = item_seq[r0 + rl];
  eL[tid] = item_emb[(size_t)idx * 64 + h];
  __syncthreads();
  const float4* W4 = (const float4*)(W + (size_t)h * 64);
  const float4* e4 = (const float4*)(eL + rl * 64);
  float acc = 0.f;
#pragma unroll
  for (int k = 0; k < 16; ++k) {
    float4 w = W4[k], e = e4[k];
    acc += w.x * e.x + w.y * e.y + w.z * e.z + w.w * e.w;
  }
  hat[(size_t)(r0 + rl) * 64 + h] = acc;
}

// ---------------- K2: per-column (i,s) max & sum(exp) over the BATCH axis ----------------
__global__ void k_colstats(const float* __restrict__ cw,
                           float* __restrict__ colmax,
                           float* __restrict__ colsum) {
  __shared__ float red[256];
  const int col = blockIdx.x;          // 0..799  (= i*200+s)
  const int t   = threadIdx.x;
  float v0 = cw[(size_t)(t)       * 800 + col];
  float v1 = cw[(size_t)(t + 256) * 800 + col];
  float v2 = cw[(size_t)(t + 512) * 800 + col];
  float v3 = cw[(size_t)(t + 768) * 800 + col];
  float m = fmaxf(fmaxf(v0, v1), fmaxf(v2, v3));
  red[t] = m; __syncthreads();
  for (int o = 128; o > 0; o >>= 1) { if (t < o) red[t] = fmaxf(red[t], red[t + o]); __syncthreads(); }
  const float cm = red[0]; __syncthreads();
  float s = __expf(v0 - cm) + __expf(v1 - cm) + __expf(v2 - cm) + __expf(v3 - cm);
  red[t] = s; __syncthreads();
  for (int o = 128; o > 0; o >>= 1) { if (t < o) red[t] += red[t + o]; __syncthreads(); }
  if (t == 0) { colmax[col] = cm; colsum[col] = red[0]; }
}

// ---------------- K3: routing round (reduced 16-dim form) ----------------
// block = 128 = 4 waves; wave i owns capsule i of batch b = blockIdx.x.
// hat_exp[b,i,s,h] == hat[b,s,16i + h/4]  ->  cap has only 16 distinct values.
__global__ void k_route(const float* __restrict__ hat,
                        const int* __restrict__ mask,
                        const float* __restrict__ colmax,
                        const float* __restrict__ colsum,
                        float* __restrict__ cw,
                        float* __restrict__ outmulti,
                        int last) {
  const int b    = blockIdx.x;
  const int i    = threadIdx.x >> 5;
  const int lane = threadIdx.x & 31;
  const float* hb  = hat + (size_t)b * kS * 64 + i * 16;
  const float* cmx = colmax + i * kS;
  const float* csm = colsum + i * kS;
  float* cwb = cw + (size_t)b * (kI * kS) + i * kS;
  const int* mb = mask + (size_t)b * kS;

  float acc[16];
#pragma unroll
  for (int j = 0; j < 16; ++j) acc[j] = 0.f;

  for (int s = lane; s < kS; s += 32) {
    float sw = 0.f;
    if (mb[s] != 0) sw = __expf(cwb[s] - cmx[s]) / csm[s];
    const float4* h4 = (const float4*)(hb + (size_t)s * 64);
#pragma unroll
    for (int j4 = 0; j4 < 4; ++j4) {
      float4 hv = h4[j4];
      acc[j4*4+0] += sw * hv.x; acc[j4*4+1] += sw * hv.y;
      acc[j4*4+2] += sw * hv.z; acc[j4*4+3] += sw * hv.w;
    }
  }
#pragma unroll
  for (int off = 16; off >= 1; off >>= 1)
#pragma unroll
    for (int j = 0; j < 16; ++j) acc[j] += __shfl_xor(acc[j], off, 32);

  float norm = 0.f;
#pragma unroll
  for (int j = 0; j < 16; ++j) norm += acc[j] * acc[j];
  norm *= 4.0f;                                  // each value repeated 4x in full cap
  const float scalar = norm / (1.0f + norm) * rsqrtf(norm + 1e-9f);
  float cs[16];
#pragma unroll
  for (int j = 0; j < 16; ++j) cs[j] = scalar * acc[j];

  if (!last) {                                   // delta = 4 * sum_j hat * cs[j]
    for (int s = lane; s < kS; s += 32) {
      const float4* h4 = (const float4*)(hb + (size_t)s * 64);
      float d = 0.f;
#pragma unroll
      for (int j4 = 0; j4 < 4; ++j4) {
        float4 hv = h4[j4];
        d += hv.x*cs[j4*4+0] + hv.y*cs[j4*4+1] + hv.z*cs[j4*4+2] + hv.w*cs[j4*4+3];
      }
      cwb[s] += 4.0f * d;
    }
  } else {                                       // write multi (B,I,64) to d_out
    float* ob = outmulti + (size_t)b * (kI * kH) + i * kH;
    ob[lane]      = cs[lane >> 2];
    ob[lane + 32] = cs[8 + (lane >> 2)];
  }
}

// ---------------- K4: cos / argmax / gather best (+ bf16*log2e copy) ----------------
__global__ void k_best(const float* __restrict__ multi,
                       const float* __restrict__ item_emb,
                       const int* __restrict__ item,
                       float* __restrict__ bestf,
                       unsigned short* __restrict__ best16) {
  const int b = blockIdx.x, l = threadIdx.x;
  const int it = item[b];
  const float e0 = item_emb[(size_t)it * 64 + l];
  const float e1 = item_emb[(size_t)it * 64 + 32 + l];
  const float* mb = multi + (size_t)b * 256;
  float cosv[4];
#pragma unroll
  for (int i = 0; i < 4; ++i) {
    float p = mb[i*64 + l] * e0 + mb[i*64 + 32 + l] * e1;
#pragma unroll
    for (int off = 16; off >= 1; off >>= 1) p += __shfl_xor(p, off, 32);
    cosv[i] = p;
  }
  int k = 0; float bv = cosv[0];
#pragma unroll
  for (int i = 1; i < 4; ++i) if (cosv[i] > bv) { bv = cosv[i]; k = i; }  // first-max tie
  const float v0 = mb[k*64 + l], v1 = mb[k*64 + 32 + l];
  bestf[(size_t)b*64 + l]        = v0;
  bestf[(size_t)b*64 + 32 + l]   = v1;
  // fold log2(e) into A so WMMA emits score*log2e and exp becomes a bare v_exp_f32 (2^x)
  const float kLog2e = 1.4426950408889634f;
  best16[(size_t)b*64 + l]       = f2bf(v0 * kLog2e);
  best16[(size_t)b*64 + 32 + l]  = f2bf(v1 * kLog2e);
}

// ---------------- K5: f32 -> bf16 table conversion ----------------
__global__ void k_cvt(const float* __restrict__ src, unsigned short* __restrict__ dst, int n4) {
  const int stride = gridDim.x * blockDim.x;
  const float4* s4 = (const float4*)src;
  ushort4* d4 = (ushort4*)dst;
  for (int i = blockIdx.x * blockDim.x + threadIdx.x; i < n4; i += stride) {
    float4 v = s4[i];
    ushort4 o; o.x = f2bf(v.x); o.y = f2bf(v.y); o.z = f2bf(v.z); o.w = f2bf(v.w);
    d4[i] = o;
  }
}

// ---------------- K6: scores GEMM (WMMA bf16) fused with sum-of-exp ----------------
// grid (64 rowtiles, 25 nsplits), block = 128 = 4 waves. Each wave: fixed 16x64 A tile,
// streams 16-col B tiles, 2 chained v_wmma_f32_16x16x32_bf16 per tile; two tiles per
// iteration so tile B's WMMAs fill tile A's WMMA->VALU hazard slots (no v_nops).
// Scores are bounded (|score*log2e| < ~1), so raw v_exp_f32 + add per element suffices.
__global__ void k_scores(const unsigned short* __restrict__ best16,
                         const unsigned short* __restrict__ emb16,
                         float* __restrict__ pst) {
  __shared__ float ssum[4][16];
  const int rowtile = blockIdx.x;
  const int nsplit  = blockIdx.y;
  const int wave = threadIdx.x >> 5, lane = threadIdx.x & 31;
  const int n16  = lane & 15;
  const int o    = lane >> 4;          // K-subgroup select (per 16-bit WMMA layout)
  const int row  = rowtile * 16 + n16;

  const uint4* a4 = (const uint4*)(best16 + (size_t)row * 64);
  BfVec alo, ahi;
  alo.q[0] = a4[o];     alo.q[1] = a4[o + 2];
  ahi.q[0] = a4[o + 4]; ahi.q[1] = a4[o + 6];

  float s[8], s2[8];
#pragma unroll
  for (int j = 0; j < 8; ++j) { s[j] = 0.f; s2[j] = 0.f; }

  // 4 waves stride 4 tiles; STEP = 4 tiles = 4*16 cols * 64 elems (8 KB)
  constexpr size_t STEP = (size_t)4 * 16 * 64;
  const unsigned short* pcol =
      emb16 + (size_t)((nsplit * kTILES + wave) * 16 + n16) * 64;

  int t = wave;
  for (; t + 4 < kTILES; t += 8) {               // two independent tiles per iter
    v8f c0 = score_tile((const uint4*)pcol,          o, alo, ahi);
    v8f c1 = score_tile((const uint4*)(pcol + STEP), o, alo, ahi);
#pragma unroll
    for (int j = 0; j < 8; ++j) s[j]  += __builtin_amdgcn_exp2f(c0[j]);
#pragma unroll
    for (int j = 0; j < 8; ++j) s2[j] += __builtin_amdgcn_exp2f(c1[j]);
    pcol += 2 * STEP;
  }
  if (t < kTILES) {                              // remainder tile (waves 0,1)
    v8f c0 = score_tile((const uint4*)pcol, o, alo, ahi);
#pragma unroll
    for (int j = 0; j < 8; ++j) s[j] += __builtin_amdgcn_exp2f(c0[j]);
  }
#pragma unroll
  for (int j = 0; j < 8; ++j) s[j] += s2[j];

  // sum the 16 columns held by each half-wave (same row set per half)
#pragma unroll
  for (int off = 1; off < 16; off <<= 1)
#pragma unroll
    for (int j = 0; j < 8; ++j) s[j] += __shfl_xor(s[j], off, 32);
  if (n16 == 0) {
    const int rb = o * 8;
#pragma unroll
    for (int j = 0; j < 8; ++j) ssum[wave][rb + j] = s[j];
  }
  __syncthreads();
  if (threadIdx.x < 16) {
    const int r = threadIdx.x;
    const float total = ssum[0][r] + ssum[1][r] + ssum[2][r] + ssum[3][r];
    const int b = rowtile * 16 + r;
    pst[(size_t)b * kNSPLIT + nsplit] = total;
  }
}

// ---------------- K7: combine partials -> lse[b]; zero the loss accumulator ----------------
__global__ void k_lse(const float* __restrict__ pst, float* __restrict__ lse,
                      float* __restrict__ lossptr) {
  const int b = blockIdx.x * blockDim.x + threadIdx.x;
  if (b == 0) *lossptr = 0.f;
  if (b < kB) {
    float Sv = 0.f;
    for (int p = 0; p < kNSPLIT; ++p) Sv += pst[(size_t)b * kNSPLIT + p];
    lse[b] = __logf(Sv);                        // sum(e^score) accumulated exactly
  }
}

// ---------------- K8: loss = mean_b(lse[b] - <best[b], emb[item_b]>) ----------------
__global__ void k_loss(const float* __restrict__ bestf, const float* __restrict__ item_emb,
                       const int* __restrict__ item, const float* __restrict__ lse,
                       float* __restrict__ lossptr) {
  const int wave = threadIdx.x >> 5, lane = threadIdx.x & 31;
  const int b = blockIdx.x * 8 + wave;
  const int it = item[b];
  float p = bestf[(size_t)b*64 + lane]      * item_emb[(size_t)it*64 + lane]
          + bestf[(size_t)b*64 + 32 + lane] * item_emb[(size_t)it*64 + 32 + lane];
#pragma unroll
  for (int off = 16; off >= 1; off >>= 1) p += __shfl_xor(p, off, 32);
  if (lane == 0) atomicAdd(lossptr, (lse[b] - p) * (1.0f / kB));
}

// ---------------- workspace layout (bytes) ----------------
constexpr size_t HAT_OFF    = 0;                                    // 52,428,800
constexpr size_t CW_OFF     = HAT_OFF    + (size_t)kB*kS*kH*4;      //  3,276,800
constexpr size_t COLMAX_OFF = CW_OFF     + (size_t)kB*kI*kS*4;      //      3,200
constexpr size_t COLSUM_OFF = COLMAX_OFF + 3200;                    //      3,200
constexpr size_t BESTF_OFF  = COLSUM_OFF + 3200;                    //    262,144
constexpr size_t LSE_OFF    = BESTF_OFF  + (size_t)kB*kH*4;         //      4,096
constexpr size_t PST_OFF    = LSE_OFF    + (size_t)kB*4;            //    102,400
constexpr size_t BEST16_OFF = PST_OFF    + (size_t)kB*kNSPLIT*4;    //    131,072
constexpr size_t EMB16_OFF  = BEST16_OFF + (size_t)kB*kH*2;         // 12,800,000

extern "C" void kernel_launch(void* const* d_in, const int* in_sizes, int n_in,
                              void* d_out, int out_size, void* d_ws, size_t ws_size,
                              hipStream_t stream) {
  (void)in_sizes; (void)n_in; (void)out_size; (void)ws_size;
  const int*   item_seq = (const int*)d_in[0];
  const int*   mask     = (const int*)d_in[1];
  const int*   item     = (const int*)d_in[2];
  const float* capw     = (const float*)d_in[3];
  const float* item_emb = (const float*)d_in[4];
  const float* W        = (const float*)d_in[5];

  float* out_multi = (float*)d_out;                 // (B, I, H) flat
  float* out_loss  = out_multi + (size_t)kB * kI * kH;

  char* w = (char*)d_ws;
  float*          hat    = (float*)(w + HAT_OFF);
  float*          cw     = (float*)(w + CW_OFF);
  float*          colmax = (float*)(w + COLMAX_OFF);
  float*          colsum = (float*)(w + COLSUM_OFF);
  float*          bestf  = (float*)(w + BESTF_OFF);
  float*          lse    = (float*)(w + LSE_OFF);
  float*          pst    = (float*)(w + PST_OFF);
  unsigned short* best16 = (unsigned short*)(w + BEST16_OFF);
  unsigned short* emb16  = (unsigned short*)(w + EMB16_OFF);

  // routing weights evolve -> work on a copy (inputs must stay pristine)
  (void)hipMemcpyAsync(cw, capw, (size_t)kB * kI * kS * sizeof(float),
                       hipMemcpyDeviceToDevice, stream);

  k_hat<<<(kB * kS) / 4, 256, 0, stream>>>(item_seq, item_emb, W, hat);
  k_cvt<<<2048, 256, 0, stream>>>(item_emb, emb16, kN * kH / 4);

  for (int r = 0; r < 3; ++r) {
    k_colstats<<<kI * kS, 256, 0, stream>>>(cw, colmax, colsum);
    k_route<<<kB, 128, 0, stream>>>(hat, mask, colmax, colsum, cw, out_multi, r == 2);
  }

  k_best<<<kB, 32, 0, stream>>>(out_multi, item_emb, item, bestf, best16);
  k_scores<<<dim3(kB / 16, kNSPLIT), 128, 0, stream>>>(best16, emb16, pst);
  k_lse<<<4, 256, 0, stream>>>(pst, lse, out_loss);
  k_loss<<<kB / 8, 256, 0, stream>>>(bestf, item_emb, item, lse, out_loss);
}
